// HGTLayer_53145925321001
// MI455X (gfx1250) — matrix operand order repaired and checked by reading, not compiled, not run on hardware
//
#include <hip/hip_runtime.h>
#include <hip/hip_bf16.h>

#define H_   8
#define DK_  64
#define DIM_ 512

typedef __attribute__((ext_vector_type(16))) __bf16         v16bf;
typedef __attribute__((ext_vector_type(16))) unsigned short v16u;
typedef __attribute__((ext_vector_type(8)))  unsigned short u16x8;
typedef __attribute__((ext_vector_type(8)))  float          v8f;

union FragU { v16u v; u16x8 h[2]; };

__device__ __forceinline__ v16bf as_bf(v16u u) { return __builtin_bit_cast(v16bf, u); }
__device__ __forceinline__ float bf2f(unsigned short b) {
    unsigned u = ((unsigned)b) << 16;
    return __builtin_bit_cast(float, u);
}
__device__ __forceinline__ unsigned short f2bf(float f) {
    unsigned u = __builtin_bit_cast(unsigned, f);
    if ((u & 0x7F800000u) == 0x7F800000u) {          // inf/NaN: preserve
        unsigned short hv = (unsigned short)(u >> 16);
        if (u & 0xFFFFu) hv |= 0x40;
        return hv;
    }
    u += 0x7FFFu + ((u >> 16) & 1u);                 // round to nearest even
    return (unsigned short)(u >> 16);
}
// order-preserving float<->uint encoding for atomicMax on floats
__device__ __forceinline__ unsigned ordEnc(float f) {
    unsigned u = __builtin_bit_cast(unsigned, f);
    return (u & 0x80000000u) ? ~u : (u | 0x80000000u);
}
__device__ __forceinline__ float ordDec(unsigned e) {
    unsigned u = (e & 0x80000000u) ? (e ^ 0x80000000u) : ~e;
    return __builtin_bit_cast(float, u);
}

// load A fragment (16x32 bf16) for this lane from a contiguous bf16 row slice
__device__ __forceinline__ v16bf load_afrag(const unsigned short* row, int k0, int lhalf) {
    FragU a;
    a.h[0] = *(const u16x8*)(row + k0 + lhalf * 8);
    a.h[1] = *(const u16x8*)(row + k0 + 16 + lhalf * 8);
    return as_bf(a.v);
}
// load B fragment (32x16 bf16) from fragment-packed B: k = 8*lhalf+0..7 and 16+8*lhalf+0..7
__device__ __forceinline__ v16bf load_bfrag(const unsigned short* packedCol, int lhalf) {
    FragU b;
    b.h[0] = *(const u16x8*)(packedCol + lhalf * 8);
    b.h[1] = *(const u16x8*)(packedCol + 16 + lhalf * 8);
    return as_bf(b.v);
}
// load the 4 B fragments (covering 64 output cols) for k-step k0
__device__ __forceinline__ void load_b4(const unsigned short* Bp, int k0, int Nn,
                                        int ncol, int lhalf, v16bf* dst) {
    const unsigned short* bb = Bp + (((size_t)(k0 >> 5) * Nn + ncol) << 5);
    #pragma unroll
    for (int b = 0; b < 4; b++) dst[b] = load_bfrag(bb + ((size_t)(b * 16) << 5), lhalf);
}

// ---------------- fp32 -> bf16 conversion (linear) ----------------
__global__ void cvt_f32_bf16(const float* __restrict__ src,
                             unsigned short* __restrict__ dst, long long n) {
    long long i = (long long)blockIdx.x * blockDim.x + threadIdx.x;
    if (i >= n) return;
    dst[i] = f2bf(src[i]);
}

// ---------------- fp32 -> bf16 + pack into WMMA B-fragment order ----------------
// src: nmat consecutive row-major [K][Nn] f32 matrices.
// dst: per matrix, element (k,n) at ((k>>5)*Nn + n)*32 + (k&31)
__global__ void pack_bf16_frag(const float* __restrict__ src,
                               unsigned short* __restrict__ dst,
                               int K, int Nn, long long total) {
    long long idx = (long long)blockIdx.x * blockDim.x + threadIdx.x;
    if (idx >= total) return;
    long long kn = (long long)K * Nn;
    int m   = (int)(idx / kn);
    int rem = (int)(idx - (long long)m * kn);
    int k = rem / Nn, n = rem - k * Nn;
    dst[(size_t)m * kn + (((size_t)(k >> 5) * Nn + n) << 5) + (k & 31)] = f2bf(src[idx]);
}

// ---------------- generic bf16 WMMA GEMM (32x64 tile/wave) ----------------
// C[M,Nn] = A[M,K](bf16,row-major) @ Bp[K,Nn](bf16,fragment-packed) + bias
// mode 0: store bf16        mode 1: store fp32
// mode 2: store fp32 with skip blend: out = (acc+bias)*sig + skipsrc*(1-sig)
__global__ void gemm16(const unsigned short* __restrict__ A,
                       const unsigned short* __restrict__ Bp,
                       const float* __restrict__ bias,
                       void* __restrict__ out,
                       const float* __restrict__ skipsrc,
                       const float* __restrict__ skipvec, int type_idx,
                       int M, int Nn, int K, int mode) {
    int lane  = threadIdx.x & 31;
    int wv    = threadIdx.x >> 5;
    int gw    = blockIdx.x * (blockDim.x >> 5) + wv;
    int mtiles = (M + 31) >> 5;          // 32-row tiles
    int ntiles = Nn >> 6;                // 64-col tiles
    if (gw >= mtiles * ntiles) return;
    int mt = gw / ntiles, nt = gw % ntiles;
    int lhalf = lane >> 4;               // 0: lanes 0-15, 1: lanes 16-31
    int mrow  = lane & 15;
    int ncol  = nt * 64 + mrow;
    int m0base = mt * 32;

    int r0 = m0base + mrow;      if (r0 >= M) r0 = M - 1;
    int r1 = m0base + 16 + mrow; if (r1 >= M) r1 = M - 1;
    const unsigned short* Arow0 = A + (size_t)r0 * K;
    const unsigned short* Arow1 = A + (size_t)r1 * K;

    v8f zf = {0.f, 0.f, 0.f, 0.f, 0.f, 0.f, 0.f, 0.f};
    v8f acc[8] = {zf, zf, zf, zf, zf, zf, zf, zf};

    for (int k0 = 0; k0 < K; k0 += 32) {
        v16bf a0 = load_afrag(Arow0, k0, lhalf);
        v16bf a1 = load_afrag(Arow1, k0, lhalf);
        v16bf bf[4];
        load_b4(Bp, k0, Nn, ncol, lhalf, bf);
        #pragma unroll
        for (int b = 0; b < 4; b++) {
            acc[b]     = __builtin_amdgcn_wmma_f32_16x16x32_bf16(
                false, a0, false, bf[b], (short)0, acc[b], false, false);
            acc[4 + b] = __builtin_amdgcn_wmma_f32_16x16x32_bf16(
                false, a1, false, bf[b], (short)0, acc[4 + b], false, false);
        }
    }

    float sig = 1.f, beta = 0.f;
    if (mode == 2) {
        float sp = skipvec[type_idx];
        sig = 1.f / (1.f + expf(-sp));
        beta = 1.f - sig;
    }
    #pragma unroll
    for (int mm = 0; mm < 2; mm++) {
        #pragma unroll
        for (int b = 0; b < 4; b++) {
            #pragma unroll
            for (int r = 0; r < 8; r++) {
                int orow = m0base + mm * 16 + r + lhalf * 8;
                if (orow >= M) continue;
                int ocol = nt * 64 + b * 16 + mrow;
                float vv = acc[mm * 4 + b][r] + bias[ocol];
                if (mode == 0) {
                    ((unsigned short*)out)[(size_t)orow * Nn + ocol] = f2bf(vv);
                } else if (mode == 1) {
                    ((float*)out)[(size_t)orow * Nn + ocol] = vv;
                } else {
                    ((float*)out)[(size_t)orow * Nn + ocol] =
                        vv * sig + skipsrc[(size_t)orow * Nn + ocol] * beta;
                }
            }
        }
    }
}

// ---------------- edge attention logits ----------------
// per wave: 16 edges x one head. key_e = k[src] @ relA[h]  (WMMA),
// att = (q[dst] . key_e) * pri / sqrt(DK); segment-max via ordered atomicMax.
__global__ void edge_att(const unsigned short* __restrict__ kb,
                         const unsigned short* __restrict__ qb,
                         const int* __restrict__ src, const int* __restrict__ dst,
                         const unsigned short* __restrict__ rel,
                         const float* __restrict__ pri,
                         float* __restrict__ att, unsigned* __restrict__ menc,
                         int E) {
    int lane = threadIdx.x & 31;
    int wv   = threadIdx.x >> 5;
    int gw   = blockIdx.x * (blockDim.x >> 5) + wv;
    int etiles = (E + 15) >> 4;
    if (gw >= etiles * H_) return;
    int h  = gw & (H_ - 1);
    int et = gw >> 3;
    int e0 = et * 16;
    int lhalf = lane >> 4;
    int mrow  = lane & 15;

    int eg = e0 + mrow;
    if (eg >= E) eg = E - 1;
    int sidx = src[eg];
    const unsigned short* arow = kb + (size_t)sidx * DIM_ + h * DK_;
    const unsigned short* relh = rel + (size_t)h * DK_ * DK_;

    // load all fragments up front (2 A, 8 B), then 8 back-to-back WMMAs
    v16bf af0 = load_afrag(arow, 0, lhalf);
    v16bf af1 = load_afrag(arow, 32, lhalf);
    v16bf bf0[4], bf1[4];
    load_b4(relh, 0, DK_, mrow, lhalf, bf0);
    load_b4(relh, 32, DK_, mrow, lhalf, bf1);

    v8f zf = {0.f, 0.f, 0.f, 0.f, 0.f, 0.f, 0.f, 0.f};
    v8f acc[4] = {zf, zf, zf, zf};
    #pragma unroll
    for (int b = 0; b < 4; b++)
        acc[b] = __builtin_amdgcn_wmma_f32_16x16x32_bf16(
            false, af0, false, bf0[b], (short)0, acc[b], false, false);
    #pragma unroll
    for (int b = 0; b < 4; b++)
        acc[b] = __builtin_amdgcn_wmma_f32_16x16x32_bf16(
            false, af1, false, bf1[b], (short)0, acc[b], false, false);

    const float inv_sqrt_dk = 0.125f;
    float pr = pri[h];
    #pragma unroll
    for (int r = 0; r < 8; r++) {
        int Mr  = r + lhalf * 8;
        int eg2 = e0 + Mr;
        int egc = (eg2 < E) ? eg2 : (E - 1);
        int dd  = dst[egc];
        const unsigned short* qrow = qb + (size_t)dd * DIM_ + h * DK_ + mrow;
        float p = 0.f;
        #pragma unroll
        for (int b = 0; b < 4; b++) p += acc[b][r] * bf2f(qrow[b * 16]);
        #pragma unroll
        for (int o = 1; o < 16; o <<= 1) p += __shfl_xor(p, o, 32);
        if (mrow == r && eg2 < E) {
            float a_ = p * pr * inv_sqrt_dk;
            att[(size_t)eg2 * H_ + h] = a_;
            atomicMax(menc + (size_t)dd * H_ + h, ordEnc(a_));
        }
    }
}

// ---------------- segment softmax normalize pass ----------------
__global__ void soft_norm(float* __restrict__ att, const unsigned* __restrict__ menc,
                          float* __restrict__ ssum, const int* __restrict__ dst, int E) {
    int idx = blockIdx.x * blockDim.x + threadIdx.x;
    if (idx >= E * H_) return;
    int e = idx >> 3, h = idx & (H_ - 1);
    int dd = dst[e];
    float mv = ordDec(menc[(size_t)dd * H_ + h]);
    float ex = expf(att[idx] - mv);
    att[idx] = ex;
    atomicAdd(ssum + (size_t)dd * H_ + h, ex);
}

// ---------------- edge aggregation ----------------
// val_e = v[src] @ relM[h] (WMMA); t[dst] += (e/s[dst]) * val_e  (atomic scatter)
__global__ void edge_agg(const unsigned short* __restrict__ vb,
                         const int* __restrict__ src, const int* __restrict__ dst,
                         const unsigned short* __restrict__ rel,
                         const float* __restrict__ att, const float* __restrict__ ssum,
                         float* __restrict__ t, int E) {
    int lane = threadIdx.x & 31;
    int wv   = threadIdx.x >> 5;
    int gw   = blockIdx.x * (blockDim.x >> 5) + wv;
    int etiles = (E + 15) >> 4;
    if (gw >= etiles * H_) return;
    int h  = gw & (H_ - 1);
    int et = gw >> 3;
    int e0 = et * 16;
    int lhalf = lane >> 4;
    int mrow  = lane & 15;

    int eg = e0 + mrow;
    if (eg >= E) eg = E - 1;
    int sidx = src[eg];
    const unsigned short* arow = vb + (size_t)sidx * DIM_ + h * DK_;
    const unsigned short* relh = rel + (size_t)h * DK_ * DK_;

    v16bf af0 = load_afrag(arow, 0, lhalf);
    v16bf af1 = load_afrag(arow, 32, lhalf);
    v16bf bf0[4], bf1[4];
    load_b4(relh, 0, DK_, mrow, lhalf, bf0);
    load_b4(relh, 32, DK_, mrow, lhalf, bf1);

    v8f zf = {0.f, 0.f, 0.f, 0.f, 0.f, 0.f, 0.f, 0.f};
    v8f acc[4] = {zf, zf, zf, zf};
    #pragma unroll
    for (int b = 0; b < 4; b++)
        acc[b] = __builtin_amdgcn_wmma_f32_16x16x32_bf16(
            false, af0, false, bf0[b], (short)0, acc[b], false, false);
    #pragma unroll
    for (int b = 0; b < 4; b++)
        acc[b] = __builtin_amdgcn_wmma_f32_16x16x32_bf16(
            false, af1, false, bf1[b], (short)0, acc[b], false, false);

    #pragma unroll
    for (int r = 0; r < 8; r++) {
        int Mr  = r + lhalf * 8;
        int eg2 = e0 + Mr;
        bool ok = eg2 < E;
        int egc = ok ? eg2 : (E - 1);
        int dd  = dst[egc];
        float ev = att[(size_t)egc * H_ + h];
        float sv = ssum[(size_t)dd * H_ + h];
        float coeff = ok ? (ev / sv) : 0.f;
        float* trow = t + (size_t)dd * DIM_ + h * DK_ + mrow;
        #pragma unroll
        for (int b = 0; b < 4; b++) atomicAdd(trow + b * 16, acc[b][r] * coeff);
    }
}

extern "C" void kernel_launch(void* const* d_in, const int* in_sizes, int n_in,
                              void* d_out, int out_size, void* d_ws, size_t ws_size,
                              hipStream_t stream) {
    (void)n_in; (void)out_size; (void)ws_size;
    const int N = in_sizes[0] / DIM_;
    const int E = in_sizes[2];

    const float* h_paper  = (const float*)d_in[0];
    const float* h_author = (const float*)d_in[1];
    const int*   src0     = (const int*)d_in[2];
    const int*   dst0     = (const int*)d_in[3];
    const int*   src1     = (const int*)d_in[4];
    const int*   dst1     = (const int*)d_in[5];
    // W order per type: [0]=Wk [1]=Wq [2]=Wv [3]=Wa
    const float* W[8]; const float* Bv[8];
    for (int t = 0; t < 2; t++)
        for (int j = 0; j < 4; j++) {
            W[t * 4 + j]  = (const float*)d_in[6 + t * 8 + j * 2];
            Bv[t * 4 + j] = (const float*)d_in[6 + t * 8 + j * 2 + 1];
        }
    const float* rel_att = (const float*)d_in[22];
    const float* rel_msg = (const float*)d_in[23];
    const float* rel_pri = (const float*)d_in[24];
    const float* skipv   = (const float*)d_in[25];
    float* out = (float*)d_out;

    // workspace carve
    char* wsp = (char*)d_ws; size_t off = 0;
    auto carve = [&](size_t bytes) -> void* {
        void* p = (void*)(wsp + off);
        off += bytes; off = (off + 255) & ~(size_t)255;
        return p;
    };
    const size_t nd = (size_t)N * DIM_;
    unsigned short* hbf0 = (unsigned short*)carve(nd * 2);
    unsigned short* hbf1 = (unsigned short*)carve(nd * 2);
    unsigned short* wbf[8];
    for (int i = 0; i < 8; i++) wbf[i] = (unsigned short*)carve((size_t)DIM_ * DIM_ * 2);
    unsigned short* relAbf = (unsigned short*)carve((size_t)2 * H_ * DK_ * DK_ * 2);
    unsigned short* relMbf = (unsigned short*)carve((size_t)2 * H_ * DK_ * DK_ * 2);
    unsigned short* kbf0 = (unsigned short*)carve(nd * 2);
    unsigned short* vbf0 = (unsigned short*)carve(nd * 2);
    unsigned short* qbf0 = (unsigned short*)carve(nd * 2);
    unsigned short* kbf1 = (unsigned short*)carve(nd * 2);
    unsigned short* vbf1 = (unsigned short*)carve(nd * 2);
    unsigned short* qbf1 = (unsigned short*)carve(nd * 2);
    float* att0 = (float*)carve((size_t)E * H_ * 4);
    float* att1 = (float*)carve((size_t)E * H_ * 4);
    unsigned* m0 = (unsigned*)carve((size_t)N * H_ * 4);
    unsigned* m1 = (unsigned*)carve((size_t)N * H_ * 4);
    float* s0 = (float*)carve((size_t)N * H_ * 4);
    float* s1 = (float*)carve((size_t)N * H_ * 4);
    float* t0 = (float*)carve(nd * 4);
    float* t1 = (float*)carve(nd * 4);
    unsigned short* tbf0 = (unsigned short*)carve(nd * 2);
    unsigned short* tbf1 = (unsigned short*)carve(nd * 2);

    dim3 blk(256);
    auto cvt = [&](const float* s_, unsigned short* d_, size_t n_) {
        int b = (int)((n_ + 255) / 256);
        cvt_f32_bf16<<<dim3(b), blk, 0, stream>>>(s_, d_, (long long)n_);
    };
    auto packB = [&](const float* s_, unsigned short* d_, int K, int Nn, int nmat) {
        long long n_ = (long long)K * Nn * nmat;
        int b = (int)((n_ + 255) / 256);
        pack_bf16_frag<<<dim3(b), blk, 0, stream>>>(s_, d_, K, Nn, n_);
    };
    auto gemmL = [&](const unsigned short* A, const unsigned short* B, const float* bias,
                     void* o, const float* skipsrc, int type_idx, int M, int Nn, int K, int mode) {
        int mtiles = (M + 31) >> 5, ntiles = Nn >> 6;
        int blocks = (mtiles * ntiles + 7) / 8;
        gemm16<<<dim3(blocks), blk, 0, stream>>>(A, B, bias, o, skipsrc, skipv, type_idx,
                                                 M, Nn, K, mode);
    };

    // 1. convert activations to bf16; pack weights / relation matrices into fragment order
    cvt(h_paper, hbf0, nd);
    cvt(h_author, hbf1, nd);
    for (int i = 0; i < 8; i++) packB(W[i], wbf[i], DIM_, DIM_, 1);
    packB(rel_att, relAbf, DK_, DK_, 2 * H_);
    packB(rel_msg, relMbf, DK_, DK_, 2 * H_);

    // 2. projections: k,v,q per node type (bf16 outputs, fp32 WMMA accumulate)
    gemmL(hbf0, wbf[0], Bv[0], kbf0, nullptr, 0, N, DIM_, DIM_, 0);
    gemmL(hbf0, wbf[2], Bv[2], vbf0, nullptr, 0, N, DIM_, DIM_, 0);
    gemmL(hbf0, wbf[1], Bv[1], qbf0, nullptr, 0, N, DIM_, DIM_, 0);
    gemmL(hbf1, wbf[4], Bv[4], kbf1, nullptr, 1, N, DIM_, DIM_, 0);
    gemmL(hbf1, wbf[6], Bv[6], vbf1, nullptr, 1, N, DIM_, DIM_, 0);
    gemmL(hbf1, wbf[5], Bv[5], qbf1, nullptr, 1, N, DIM_, DIM_, 0);

    // 3. zero segment state (every call: launch must be re-entrant)
    hipMemsetAsync(m0, 0, (size_t)N * H_ * 4, stream);
    hipMemsetAsync(m1, 0, (size_t)N * H_ * 4, stream);
    hipMemsetAsync(s0, 0, (size_t)N * H_ * 4, stream);
    hipMemsetAsync(s1, 0, (size_t)N * H_ * 4, stream);
    hipMemsetAsync(t0, 0, nd * 4, stream);
    hipMemsetAsync(t1, 0, nd * 4, stream);

    // 4. edge attention logits + segment max
    int etiles = (E + 15) >> 4;
    int eblocks = (etiles * H_ + 7) / 8;
    edge_att<<<dim3(eblocks), blk, 0, stream>>>(kbf1, qbf0, src0, dst0,
                                                relAbf, rel_pri, att0, m0, E);
    edge_att<<<dim3(eblocks), blk, 0, stream>>>(kbf0, qbf1, src1, dst1,
                                                relAbf + (size_t)H_ * DK_ * DK_,
                                                rel_pri + H_, att1, m1, E);

    // 5. softmax: exp + segment sum
    int sblocks = (E * H_ + 255) / 256;
    soft_norm<<<dim3(sblocks), blk, 0, stream>>>(att0, m0, s0, dst0, E);
    soft_norm<<<dim3(sblocks), blk, 0, stream>>>(att1, m1, s1, dst1, E);

    // 6. aggregate messages into t
    edge_agg<<<dim3(eblocks), blk, 0, stream>>>(vbf1, src0, dst0, relMbf,
                                                att0, s0, t0, E);
    edge_agg<<<dim3(eblocks), blk, 0, stream>>>(vbf0, src1, dst1,
                                                relMbf + (size_t)H_ * DK_ * DK_,
                                                att1, s1, t1, E);

    // 7. final output GEMM with sigmoid skip blend
    cvt(t0, tbf0, nd);
    cvt(t1, tbf1, nd);
    gemmL(tbf0, wbf[3], Bv[3], out,      h_paper,  0, N, DIM_, DIM_, 2);
    gemmL(tbf1, wbf[7], Bv[7], out + nd, h_author, 1, N, DIM_, DIM_, 2);
}
